// SwinTransformerBlock_71949292143066
// MI455X (gfx1250) — matrix operand order repaired and checked by reading, not compiled, not run on hardware
//
#include <hip/hip_runtime.h>
#include <cstdint>

#define SHIFT_ 3
#define NTOK 200704          // 64 * 56 * 56
#define SCALE_Q 0.17677669529663687f   // 32^-0.5

typedef __bf16 v16bf __attribute__((ext_vector_type(16)));
typedef float  v8f   __attribute__((ext_vector_type(8)));
typedef unsigned int u32x4 __attribute__((ext_vector_type(4)));
typedef int          i32x4 __attribute__((ext_vector_type(4)));
typedef int          i32x8 __attribute__((ext_vector_type(8)));

union Frag {
  v16bf v;
  uint4 q[2];
  unsigned short e[16];
};

__device__ __forceinline__ unsigned short f2bf(float f) {
  unsigned u = __float_as_uint(f);
  u += 0x7FFFu + ((u >> 16) & 1u);          // round-to-nearest-even
  return (unsigned short)(u >> 16);
}

__device__ __forceinline__ v8f wmma_bf16(const Frag& a, const Frag& b, v8f c) {
  return __builtin_amdgcn_wmma_f32_16x16x32_bf16(false, a.v, false, b.v, (short)0, c,
                                                 false, false);
}

// TDM: DMA a [rows x 128] bf16 tile (row-major, stride 128) from global into LDS.
// D# built per CDNA5 ISA 8.3/8.4: group0 {count=1, lds_addr, global_addr, type=2},
// group1 {data_size=2B, tensor==tile (no OOB), tile 128 x rows, stride0=128}.
__device__ __forceinline__ void tdm_load_tile_128(const unsigned short* gsrc,
                                                  unsigned short* lds_dst,
                                                  unsigned rows) {
  unsigned long long ga = (unsigned long long)(uintptr_t)gsrc;
  unsigned lds_off = (unsigned)(uintptr_t)lds_dst;   // flat LDS addr: low 32 = offset
  u32x4 g0 = { 1u,                                   // count=1, user desc
               lds_off,
               (unsigned)ga,
               (unsigned)(ga >> 32) | 0x80000000u }; // global_addr[56:32] | type=2<<30
  i32x8 g1 = { 0x00010000,                           // data_size=1 (2 bytes)
               (int)(128u << 16),                    // tensor_dim0 = 128
               (int)(rows << 16),                    // tensor_dim1 = rows
               (int)(128u << 16),                    // tile_dim0 = 128
               (int)rows,                            // tile_dim1 = rows, tile_dim2 = 0
               128,                                  // tensor_dim0_stride = 128
               0, 0 };
  i32x4 gz = { 0, 0, 0, 0 };
#if defined(__clang_major__) && (__clang_major__ >= 23)
  i32x8 gz8 = { 0, 0, 0, 0, 0, 0, 0, 0 };
  __builtin_amdgcn_tensor_load_to_lds(g0, g1, gz, gz, gz8, 0);
#else
  __builtin_amdgcn_tensor_load_to_lds(g0, g1, gz, gz, 0);
#endif
}

// ---------------------------------------------------------------- weights -> bf16
__global__ void cvt_weights_kernel(const float* __restrict__ wq,
                                   const float* __restrict__ wp,
                                   const float* __restrict__ w1,
                                   const float* __restrict__ w2,
                                   unsigned short* __restrict__ out) {
  int i = blockIdx.x * 256 + threadIdx.x;
  float v;
  if (i < 49152)        v = wq[i];
  else if (i < 65536)   v = wp[i - 49152];
  else if (i < 131072)  v = w1[i - 65536];
  else if (i < 196608)  v = w2[i - 131072];
  else return;
  out[i] = f2bf(v);
}

// ---------------------------------------------------------------- layernorm (C=128)
__global__ __launch_bounds__(256) void layernorm_kernel(const float* __restrict__ x,
                                                        const float* __restrict__ g,
                                                        const float* __restrict__ b,
                                                        unsigned short* __restrict__ out) {
  int wv = threadIdx.x >> 5;
  int lane = threadIdx.x & 31;
  long tok = (long)blockIdx.x * 8 + wv;
  const float* px = x + tok * 128;
  float v[4], s = 0.f;
#pragma unroll
  for (int i = 0; i < 4; ++i) { v[i] = px[lane + 32 * i]; s += v[i]; }
#pragma unroll
  for (int off = 16; off > 0; off >>= 1) s += __shfl_xor(s, off, 32);
  float mu = s * (1.f / 128.f);
  float var = 0.f;
#pragma unroll
  for (int i = 0; i < 4; ++i) { float d = v[i] - mu; var += d * d; }
#pragma unroll
  for (int off = 16; off > 0; off >>= 1) var += __shfl_xor(var, off, 32);
  float rs = rsqrtf(var * (1.f / 128.f) + 1e-5f);
  unsigned short* po = out + tok * 128;
#pragma unroll
  for (int i = 0; i < 4; ++i) {
    int c = lane + 32 * i;
    po[c] = f2bf((v[i] - mu) * rs * g[c] + b[c]);
  }
}

// ---------------------------------------------------------------- generic GEMM
// D[M x NOUT] = A[M x K] * W^T (+bias, +epilogue).  W row-major [NOUT x K] == B col-major.
// block: 256 thr = 8 waves (2 x 4), block tile 64 rows x 128 cols; wave tile 32x32.
// A block tile (64 x 128 bf16 = 16KB) staged in LDS via one TDM DMA per block.
template <int KDIM, int EPI>
__global__ __launch_bounds__(256) void gemm_kernel(const unsigned short* __restrict__ A,
                                                   const unsigned short* __restrict__ B,
                                                   const float* __restrict__ bias,
                                                   const float* __restrict__ resid,
                                                   unsigned short* __restrict__ outBf,
                                                   float* __restrict__ outF,
                                                   int NOUT) {
  __shared__ unsigned short sA[64 * KDIM];
  int wid = threadIdx.x >> 5;
  int lane = threadIdx.x & 31;
  int hv = lane >> 4, lr = lane & 15;
  long rowBase = (long)blockIdx.x * 64 + (wid >> 2) * 32;
  int rowLocal = (wid >> 2) * 32;
  int colBase = blockIdx.y * 128 + (wid & 3) * 32;
  __builtin_prefetch(B + (size_t)colBase * KDIM, 0, 0);
  if (wid == 0) {  // one TDM per workgroup: A tile -> LDS, tracked by TENSORcnt
    tdm_load_tile_128(A + (size_t)blockIdx.x * 64 * KDIM, sA, 64);
    __builtin_amdgcn_s_wait_tensorcnt(0);
  }
  __syncthreads();
  v8f acc[2][2] = {};
#pragma unroll
  for (int k0 = 0; k0 < KDIM; k0 += 32) {
    Frag a[2], bb[2];
#pragma unroll
    for (int i = 0; i < 2; ++i) {
      const unsigned short* pa = sA + (rowLocal + i * 16 + lr) * KDIM + k0 + hv * 16;
      a[i].q[0] = *(const uint4*)pa;
      a[i].q[1] = *(const uint4*)(pa + 8);
    }
#pragma unroll
    for (int j = 0; j < 2; ++j) {
      const unsigned short* pb = B + (size_t)(colBase + j * 16 + lr) * KDIM + k0 + hv * 16;
      bb[j].q[0] = *(const uint4*)pb;
      bb[j].q[1] = *(const uint4*)(pb + 8);
    }
#pragma unroll
    for (int i = 0; i < 2; ++i)
#pragma unroll
      for (int j = 0; j < 2; ++j) acc[i][j] = wmma_bf16(a[i], bb[j], acc[i][j]);
  }
#pragma unroll
  for (int i = 0; i < 2; ++i)
#pragma unroll
    for (int j = 0; j < 2; ++j) {
      int col = colBase + j * 16 + lr;
      float bv = bias[col];
#pragma unroll
      for (int g = 0; g < 8; ++g) {
        long row = rowBase + i * 16 + hv * 8 + g;
        size_t idx = (size_t)row * NOUT + col;
        float v = acc[i][j][g] + bv;
        if constexpr (EPI == 0) {          // QKV: pre-scale q block, store bf16
          if (col < 128) v *= SCALE_Q;
          outBf[idx] = f2bf(v);
        } else {                            // proj: + fp32 residual, store fp32
          outF[idx] = v + resid[idx];
        }
      }
    }
}

// ---------------------------------------------------------------- windowed attention
// one block per window (4096), 4 waves = 4 heads. N=49 padded to 64, HD=32.
__global__ __launch_bounds__(128) void attn_kernel(const unsigned short* __restrict__ qkv,
                                                   const float* __restrict__ rpb,
                                                   const float* __restrict__ mask,
                                                   unsigned short* __restrict__ outp) {
  __shared__ unsigned short smem[3 * 8192];  // sQ|sK|sV ; sP aliases sQ+sK after barrier
  unsigned short* sQ = smem;
  unsigned short* sK = smem + 8192;
  unsigned short* sV = smem + 16384;
  int wid = threadIdx.x >> 5;
  int lane = threadIdx.x & 31;
  int hv = lane >> 4, lr = lane & 15;
  int w = blockIdx.x;
  int bimg = w >> 6;
  int wi = w & 63;
  int wh = wi >> 3, ww = wi & 7;

  // gather shifted-window Q/K/V rows into LDS (rows 49..63 zero)
  for (int c = threadIdx.x; c < 1024; c += 128) {
    int row = c >> 4, part = c & 15;
    uint4 zq = make_uint4(0, 0, 0, 0), zk = zq, zv = zq;
    if (row < 49) {
      int rh = row / 7, rw = row - rh * 7;
      int hh = (wh * 7 + rh + SHIFT_) % 56;
      int wc = (ww * 7 + rw + SHIFT_) % 56;
      long tok = (long)bimg * 3136 + hh * 56 + wc;
      const unsigned short* p = qkv + tok * 384 + part * 8;
      zq = *(const uint4*)p;
      zk = *(const uint4*)(p + 128);
      zv = *(const uint4*)(p + 256);
    }
    *(uint4*)&sQ[row * 128 + part * 8] = zq;
    *(uint4*)&sK[row * 128 + part * 8] = zk;
    *(uint4*)&sV[row * 128 + part * 8] = zv;
  }
  __syncthreads();

  int hd0 = wid * 32;
  // S = Q @ K^T   (64x64, one K-step since HD==32)
  v8f s[4][4] = {};
#pragma unroll
  for (int ti = 0; ti < 4; ++ti) {
    Frag a;
    const unsigned short* pa = sQ + (ti * 16 + lr) * 128 + hd0 + hv * 16;
    a.q[0] = *(const uint4*)pa;
    a.q[1] = *(const uint4*)(pa + 8);
#pragma unroll
    for (int tj = 0; tj < 4; ++tj) {
      Frag bb;
      const unsigned short* pb = sK + (tj * 16 + lr) * 128 + hd0 + hv * 16;
      bb.q[0] = *(const uint4*)pb;
      bb.q[1] = *(const uint4*)(pb + 8);
      s[ti][tj] = wmma_bf16(a, bb, s[ti][tj]);
    }
  }
  __syncthreads();  // everyone done with sQ/sK -> reuse as per-wave P
  unsigned short* sP = smem + wid * 4096;  // 64x64 bf16 per wave
  const float* maskw = mask + wi * 2401;

  // bias + mask + softmax, write P (bf16) to LDS
#pragma unroll
  for (int ti = 0; ti < 4; ++ti) {
#pragma unroll
    for (int g = 0; g < 8; ++g) {
      int qrow = ti * 16 + hv * 8 + g;
      int qi = qrow / 7, qj = qrow - qi * 7;
      float vals[4];
      float mx = -1e30f;
#pragma unroll
      for (int tj = 0; tj < 4; ++tj) {
        int kc = tj * 16 + lr;
        float sv = s[ti][tj][g];
        if (kc < 49) {
          if (qrow < 49) {
            int ki = kc / 7, kj = kc - ki * 7;
            int rel = (qi - ki + 6) * 13 + (qj - kj + 6);
            sv += rpb[rel * 4 + wid] + maskw[qrow * 49 + kc];
          } else {
            sv = 0.f;                       // padded query rows: uniform, never stored
          }
        } else {
          sv = -1e30f;                      // padded key cols -> exp = 0
        }
        vals[tj] = sv;
        mx = fmaxf(mx, sv);
      }
#pragma unroll
      for (int off = 1; off < 16; off <<= 1) mx = fmaxf(mx, __shfl_xor(mx, off, 32));
      float sum = 0.f;
#pragma unroll
      for (int tj = 0; tj < 4; ++tj) { vals[tj] = __expf(vals[tj] - mx); sum += vals[tj]; }
#pragma unroll
      for (int off = 1; off < 16; off <<= 1) sum += __shfl_xor(sum, off, 32);
      float inv = 1.f / sum;
#pragma unroll
      for (int tj = 0; tj < 4; ++tj) sP[qrow * 64 + tj * 16 + lr] = f2bf(vals[tj] * inv);
    }
  }

  // O = P @ V  (64x32, K=64 in two steps)
  v8f o[4][2] = {};
#pragma unroll
  for (int kk = 0; kk < 64; kk += 32) {
    Frag bb[2];
#pragma unroll
    for (int tj = 0; tj < 2; ++tj) {
#pragma unroll
      for (int t = 0; t < 16; ++t)
        bb[tj].e[t] = sV[(kk + hv * 16 + t) * 128 + hd0 + tj * 16 + lr];
    }
#pragma unroll
    for (int ti = 0; ti < 4; ++ti) {
      Frag a;
      const unsigned short* pa = sP + (ti * 16 + lr) * 64 + kk + hv * 16;
      a.q[0] = *(const uint4*)pa;
      a.q[1] = *(const uint4*)(pa + 8);
#pragma unroll
      for (int tj = 0; tj < 2; ++tj) o[ti][tj] = wmma_bf16(a, bb[tj], o[ti][tj]);
    }
  }

  // scatter back to token order (inverse shift == same map)
#pragma unroll
  for (int ti = 0; ti < 4; ++ti)
#pragma unroll
    for (int tj = 0; tj < 2; ++tj)
#pragma unroll
      for (int g = 0; g < 8; ++g) {
        int qrow = ti * 16 + hv * 8 + g;
        if (qrow < 49) {
          int rh = qrow / 7, rw = qrow - rh * 7;
          int hh = (wh * 7 + rh + SHIFT_) % 56;
          int wc = (ww * 7 + rw + SHIFT_) % 56;
          long tok = (long)bimg * 3136 + hh * 56 + wc;
          outp[tok * 128 + hd0 + tj * 16 + lr] = f2bf(o[ti][tj][g]);
        }
      }
}

// ---------------------------------------------------------------- fused MLP
// per 64-row block: h = gelu(Xn @ W1^T + b1) staged in LDS, out += h @ W2^T + b2 (in place)
__global__ __launch_bounds__(256) void mlp_kernel(const unsigned short* __restrict__ Xn,
                                                  const unsigned short* __restrict__ W1,
                                                  const float* __restrict__ B1,
                                                  const unsigned short* __restrict__ W2,
                                                  const float* __restrict__ B2,
                                                  float* __restrict__ xio) {
  __shared__ unsigned short sH[64 * 512];
  int wid = threadIdx.x >> 5;
  int lane = threadIdx.x & 31;
  int hv = lane >> 4, lr = lane & 15;
  int wr = wid >> 2, wc = wid & 3;
  long rowBase = (long)blockIdx.x * 64 + wr * 32;

  {  // phase 1: fc1 + gelu -> sH, wave tile 32 x 128
    v8f acc[2][8] = {};
#pragma unroll
    for (int k0 = 0; k0 < 128; k0 += 32) {
      Frag a[2];
#pragma unroll
      for (int i = 0; i < 2; ++i) {
        const unsigned short* pa = Xn + (rowBase + i * 16 + lr) * 128 + k0 + hv * 16;
        a[i].q[0] = *(const uint4*)pa;
        a[i].q[1] = *(const uint4*)(pa + 8);
      }
#pragma unroll
      for (int j = 0; j < 8; ++j) {
        Frag bb;
        const unsigned short* pb = W1 + (size_t)(wc * 128 + j * 16 + lr) * 128 + k0 + hv * 16;
        bb.q[0] = *(const uint4*)pb;
        bb.q[1] = *(const uint4*)(pb + 8);
#pragma unroll
        for (int i = 0; i < 2; ++i) acc[i][j] = wmma_bf16(a[i], bb, acc[i][j]);
      }
    }
#pragma unroll
    for (int i = 0; i < 2; ++i)
#pragma unroll
      for (int j = 0; j < 8; ++j) {
        int col = wc * 128 + j * 16 + lr;
        float bv = B1[col];
#pragma unroll
        for (int g = 0; g < 8; ++g) {
          int rl = wr * 32 + i * 16 + hv * 8 + g;
          float v = acc[i][j][g] + bv;
          v = 0.5f * v * (1.f + erff(v * 0.7071067811865475f));  // exact gelu
          sH[rl * 512 + col] = f2bf(v);
        }
      }
  }
  __syncthreads();
  {  // phase 2: fc2 + residual (in place), wave tile 32 x 32, K = 512
    v8f acc[2][2] = {};
#pragma unroll
    for (int k0 = 0; k0 < 512; k0 += 32) {
      Frag a[2], bb[2];
#pragma unroll
      for (int i = 0; i < 2; ++i) {
        const unsigned short* pa = sH + (wr * 32 + i * 16 + lr) * 512 + k0 + hv * 16;
        a[i].q[0] = *(const uint4*)pa;
        a[i].q[1] = *(const uint4*)(pa + 8);
      }
#pragma unroll
      for (int j = 0; j < 2; ++j) {
        const unsigned short* pb = W2 + (size_t)(wc * 32 + j * 16 + lr) * 512 + k0 + hv * 16;
        bb[j].q[0] = *(const uint4*)pb;
        bb[j].q[1] = *(const uint4*)(pb + 8);
      }
#pragma unroll
      for (int i = 0; i < 2; ++i)
#pragma unroll
        for (int j = 0; j < 2; ++j) acc[i][j] = wmma_bf16(a[i], bb[j], acc[i][j]);
    }
#pragma unroll
    for (int i = 0; i < 2; ++i)
#pragma unroll
      for (int j = 0; j < 2; ++j) {
        int col = wc * 32 + j * 16 + lr;
        float bv = B2[col];
#pragma unroll
        for (int g = 0; g < 8; ++g) {
          long row = rowBase + i * 16 + hv * 8 + g;
          size_t idx = (size_t)row * 128 + col;
          xio[idx] = xio[idx] + acc[i][j][g] + bv;
        }
      }
  }
}

// ----------------------------------------------------------------
extern "C" void kernel_launch(void* const* d_in, const int* in_sizes, int n_in,
                              void* d_out, int out_size, void* d_ws, size_t ws_size,
                              hipStream_t stream) {
  const float* x      = (const float*)d_in[0];
  const float* amask  = (const float*)d_in[1];
  const float* n1g    = (const float*)d_in[2];
  const float* n1b    = (const float*)d_in[3];
  const float* qkv_w  = (const float*)d_in[4];
  const float* qkv_b  = (const float*)d_in[5];
  const float* rpb    = (const float*)d_in[6];
  const float* proj_w = (const float*)d_in[7];
  const float* proj_b = (const float*)d_in[8];
  const float* n2g    = (const float*)d_in[9];
  const float* n2b    = (const float*)d_in[10];
  const float* fc1_w  = (const float*)d_in[11];
  const float* fc1_b  = (const float*)d_in[12];
  const float* fc2_w  = (const float*)d_in[13];
  const float* fc2_b  = (const float*)d_in[14];
  float* outF = (float*)d_out;

  unsigned short* wBase = (unsigned short*)d_ws;
  unsigned short* wq = wBase;           // 384x128
  unsigned short* wp = wBase + 49152;   // 128x128
  unsigned short* w1 = wBase + 65536;   // 512x128
  unsigned short* w2 = wBase + 131072;  // 128x512
  unsigned short* bufA = (unsigned short*)((char*)d_ws + 393216);                       // [M x 128] bf16
  unsigned short* bufB = (unsigned short*)((char*)d_ws + 393216 + (size_t)NTOK * 256);  // [M x 384] bf16

  cvt_weights_kernel<<<768, 256, 0, stream>>>(qkv_w, proj_w, fc1_w, fc2_w, wBase);
  layernorm_kernel<<<NTOK / 8, 256, 0, stream>>>(x, n1g, n1b, bufA);
  gemm_kernel<128, 0><<<dim3(NTOK / 64, 3), 256, 0, stream>>>(bufA, wq, qkv_b, nullptr,
                                                              bufB, nullptr, 384);
  attn_kernel<<<4096, 128, 0, stream>>>(bufB, rpb, amask, bufA);
  gemm_kernel<128, 1><<<dim3(NTOK / 64, 1), 256, 0, stream>>>(bufA, wp, proj_b, x,
                                                              nullptr, outF, 128);
  layernorm_kernel<<<NTOK / 8, 256, 0, stream>>>(outF, n2g, n2b, bufA);
  mlp_kernel<<<NTOK / 64, 256, 0, stream>>>(bufA, w1, fc1_b, w2, fc2_b, outF);
}
